// SHGNN_4398046511526
// MI455X (gfx1250) — matrix operand order repaired and checked by reading, not compiled, not run on hardware
//
#include <hip/hip_runtime.h>
#include <hip/hip_bf16.h>
#include <math.h>

// ---------------- constants from the reference ----------------
#define NN 50000
#define EE 25000
#define DD 128
#define HH 4
#define CC 32
#define LL 2
#define NCLS 10
#define PAIRS 16   // pairs per segment (P_N2E/E == P_E2N/N == 16)

typedef __attribute__((ext_vector_type(16))) __bf16 v16bf;
typedef __attribute__((ext_vector_type(8)))  __bf16 v8bf;
typedef __attribute__((ext_vector_type(8)))  float  v8f;

// ---------------- weight prep: f32 [k][n] -> bf16 transposed [n][k] --------
__global__ __launch_bounds__(256)
void prep_weight(const float* __restrict__ W, __bf16* __restrict__ Wt) {
  int i = blockIdx.x * blockDim.x + threadIdx.x;   // 16384
  if (i >= DD * DD) return;
  int k = i >> 7, n = i & (DD - 1);
  Wt[(size_t)n * DD + k] = (__bf16)W[(size_t)k * DD + n];
}

// ---------------- eval-mode BatchNorm -> bf16 activations ------------------
__global__ void bn_kernel(const float* __restrict__ x,
                          const float* __restrict__ mean, const float* __restrict__ var,
                          const float* __restrict__ g, const float* __restrict__ b,
                          __bf16* __restrict__ y, int total) {
  int i = blockIdx.x * blockDim.x + threadIdx.x;
  if (i >= total) return;
  int d = i & (DD - 1);
  y[i] = (__bf16)((x[i] - mean[d]) * rsqrtf(var[d] + 1e-5f) * g[d] + b[d]);
}

// ---------------- WMMA GEMM: Y[R,128] = act(X[R,128] @ W[128,128] + bias) ----
// Xh: bf16 row-major. Wt: bf16 [n][k] (transposed). 8 waves/block, each wave
// one 16x16 tile; A tile staged in LDS via CDNA5 async global->LDS DMA.
template <int RELU>
__global__ __launch_bounds__(256)
void gemm128_bf16_wmma(const __bf16* __restrict__ Xh, const __bf16* __restrict__ Wt,
                       const float* __restrict__ bias, float* __restrict__ Y,
                       int R) {
  __shared__ __bf16 sA[16 * DD];                 // 4 KB A tile
  const int tid = threadIdx.x;
  const int row_base = blockIdx.x * 16;

  // cooperative A-tile stage: 2048 bf16 = 256 lanes x 16B, straight to LDS
  {
    const int idx = tid * 8;
    const int r = idx >> 7;                      // 0..15
    const int k = idx & (DD - 1);
    int gr = row_base + r; if (gr >= R) gr = R - 1;   // clamp (stores predicated)
    const __bf16* gsrc = Xh + (size_t)gr * DD + k;
    const unsigned lds_off = (unsigned)(size_t)(sA + idx);  // low 32b = LDS offset
    asm volatile("global_load_async_to_lds_b128 %0, %1, off"
                 :: "v"(lds_off), "v"(gsrc) : "memory");
    asm volatile("s_wait_asynccnt 0" ::: "memory");
  }
  __syncthreads();

  const int lane = tid & 31;
  const int wave = tid >> 5;
  const int half = lane >> 4;                    // 0: lanes 0-15, 1: lanes 16-31
  const int l16  = lane & 15;
  const int ncol = wave * 16 + l16;              // B column this lane supplies
  const __bf16* wrow = Wt + (size_t)ncol * DD;   // contiguous K for this column
  const __bf16* arow = sA + l16 * DD;            // A row this lane supplies

  v8f acc = {};
  #pragma unroll
  for (int k0 = 0; k0 < DD; k0 += 32) {
    // A 16x32 bf16 layout: lanes 0-15 K=k0..k0+15, lanes 16-31 K=k0+16..k0+31
    const int ka1 = k0 + (half ? 16 : 0);
    const int ka2 = k0 + (half ? 24 : 8);
    v8bf alo = *(const v8bf*)(arow + ka1);
    v8bf ahi = *(const v8bf*)(arow + ka2);
    v16bf a = __builtin_shufflevector(alo, ahi,
                                      0, 1, 2, 3, 4, 5, 6, 7,
                                      8, 9, 10, 11, 12, 13, 14, 15);
    // B 32x16 bf16 layout: lanes 0-15 K=k0..k0+15, lanes 16-31 K=k0+16..k0+31
    const int kb = k0 + (half ? 16 : 0);
    v16bf b = *(const v16bf*)(wrow + kb);
    acc = __builtin_amdgcn_wmma_f32_16x16x32_bf16(false, a, false, b,
                                                  (short)0, acc, false, false);
  }

  // epilogue: uniform full-tile fast path (R%16==0 for all but E's last block)
  const float bv = bias[ncol];
  float* yrow = Y + (size_t)(row_base + half * 8) * DD + ncol;
  if (row_base + 16 <= R) {
    #pragma unroll
    for (int r = 0; r < 8; ++r) {                // C/D: VGPR r -> M=r / M=r+8
      float y = acc[r] + bv;
      if (RELU) y = fmaxf(y, 0.f);
      yrow[(size_t)r * DD] = y;
    }
  } else {
    #pragma unroll
    for (int r = 0; r < 8; ++r) {
      if (row_base + half * 8 + r < R) {
        float y = acc[r] + bv;
        if (RELU) y = fmaxf(y, 0.f);
        yrow[(size_t)r * DD] = y;
      }
    }
  }
}

// ---------------- PMA pooling: one block (128 thr) per segment -------------
// Segments are exactly 16 contiguous pairs. Fuses: seed-attention score,
// per-head softmax, weighted V-sum, +att_r, LayerNorm(g0,b0). Emits f32
// (residual path) and bf16 (FF GEMM input).
__global__ __launch_bounds__(128)
void pma_pool_ln(const float* __restrict__ Kf, const float* __restrict__ Vf,
                 const int* __restrict__ src, const float* __restrict__ att_r,
                 const float* __restrict__ g0, const float* __restrict__ b0,
                 float* __restrict__ outf, __bf16* __restrict__ outh, int nseg) {
  __shared__ float s_alpha[HH * PAIRS];
  __shared__ float s_score[HH * PAIRS];
  __shared__ int   s_src[PAIRS];
  __shared__ float s_red[DD];

  const int seg = blockIdx.x;
  const int t = threadIdx.x;
  const int base = seg * PAIRS;

  if (t < PAIRS) s_src[t] = src[base + t];
  __syncthreads();

  // scores: 16 pairs x 4 heads = 64 dot products of length 32
  if (t < HH * PAIRS) {
    const int j = t & 15, h = t >> 4;
    const float* krow = Kf + (size_t)s_src[j] * DD + h * CC;
    const float* ar = att_r + h * CC;
    float s = 0.f;
    #pragma unroll
    for (int c = 0; c < CC; ++c) s = fmaf(krow[c], ar[c], s);
    s_score[h * PAIRS + j] = (s > 0.f) ? s : 0.2f * s;   // leaky_relu(0.2)
  }
  __syncthreads();

  // per-head softmax over the 16 pairs
  if (t < HH) {
    float m = -INFINITY;
    #pragma unroll
    for (int j = 0; j < PAIRS; ++j) m = fmaxf(m, s_score[t * PAIRS + j]);
    float den = 0.f;
    #pragma unroll
    for (int j = 0; j < PAIRS; ++j) {
      float e = __expf(s_score[t * PAIRS + j] - m);
      s_alpha[t * PAIRS + j] = e;
      den += e;
    }
    const float inv = 1.f / den;
    #pragma unroll
    for (int j = 0; j < PAIRS; ++j) s_alpha[t * PAIRS + j] *= inv;
  }
  __syncthreads();

  // weighted sum of V rows; thread t owns channel d=t
  const int d = t, h = d >> 5;
  float acc = 0.f;
  #pragma unroll
  for (int j = 0; j < PAIRS; ++j)
    acc = fmaf(s_alpha[h * PAIRS + j], Vf[(size_t)s_src[j] * DD + d], acc);
  const float val = acc + att_r[d];

  // LayerNorm over 128 channels
  s_red[t] = val; __syncthreads();
  for (int off = 64; off > 0; off >>= 1) {
    if (t < off) s_red[t] += s_red[t + off];
    __syncthreads();
  }
  const float mean = s_red[0] * (1.f / DD);
  __syncthreads();
  const float dv = val - mean;
  s_red[t] = dv * dv; __syncthreads();
  for (int off = 64; off > 0; off >>= 1) {
    if (t < off) s_red[t] += s_red[t + off];
    __syncthreads();
  }
  const float var = s_red[0] * (1.f / DD);
  const float y = dv * rsqrtf(var + 1e-5f) * g0[d] + b0[d];
  outf[(size_t)seg * DD + d] = y;
  outh[(size_t)seg * DD + d] = (__bf16)y;
}

// ---------------- residual + LayerNorm(g1,b1) + ELU ------------------------
// outf nullable (edge features only feed GEMMs); outh always written.
__global__ __launch_bounds__(128)
void resid_ln_elu(const float* __restrict__ T0, const float* __restrict__ F,
                  const float* __restrict__ g1, const float* __restrict__ b1,
                  float* __restrict__ outf, __bf16* __restrict__ outh, int R) {
  __shared__ float s_red[DD];
  const int row = blockIdx.x, t = threadIdx.x;
  const size_t base = (size_t)row * DD;
  const float val = T0[base + t] + F[base + t];

  s_red[t] = val; __syncthreads();
  for (int off = 64; off > 0; off >>= 1) {
    if (t < off) s_red[t] += s_red[t + off];
    __syncthreads();
  }
  const float mean = s_red[0] * (1.f / DD);
  __syncthreads();
  const float dv = val - mean;
  s_red[t] = dv * dv; __syncthreads();
  for (int off = 64; off > 0; off >>= 1) {
    if (t < off) s_red[t] += s_red[t + off];
    __syncthreads();
  }
  const float var = s_red[0] * (1.f / DD);
  const float y = dv * rsqrtf(var + 1e-5f) * g1[t] + b1[t];
  const float z = (y > 0.f) ? y : expm1f(y);             // ELU(alpha=1)
  if (outf) outf[base + t] = z;
  outh[base + t] = (__bf16)z;
}

// ---------------- classifier: concat -> [256,10] GEMV -> log_softmax -------
__global__ __launch_bounds__(128)
void classifier(const float* __restrict__ r0, const float* __restrict__ r1,
                const float* __restrict__ Wc, const float* __restrict__ bc,
                float* __restrict__ out, int n) {
  __shared__ float sW[LL * DD * NCLS];
  __shared__ float sb[NCLS];
  for (int i = threadIdx.x; i < LL * DD * NCLS; i += blockDim.x) sW[i] = Wc[i];
  if (threadIdx.x < NCLS) sb[threadIdx.x] = bc[threadIdx.x];
  __syncthreads();

  const int node = blockIdx.x * blockDim.x + threadIdx.x;
  if (node >= n) return;
  float logit[NCLS];
  #pragma unroll
  for (int c = 0; c < NCLS; ++c) logit[c] = sb[c];
  const float* a0 = r0 + (size_t)node * DD;
  const float* a1 = r1 + (size_t)node * DD;
  for (int k = 0; k < DD; ++k) {
    const float av = a0[k];
    #pragma unroll
    for (int c = 0; c < NCLS; ++c) logit[c] = fmaf(av, sW[k * NCLS + c], logit[c]);
  }
  for (int k = 0; k < DD; ++k) {
    const float av = a1[k];
    #pragma unroll
    for (int c = 0; c < NCLS; ++c) logit[c] = fmaf(av, sW[(DD + k) * NCLS + c], logit[c]);
  }
  float m = logit[0];
  #pragma unroll
  for (int c = 1; c < NCLS; ++c) m = fmaxf(m, logit[c]);
  float lse = 0.f;
  #pragma unroll
  for (int c = 0; c < NCLS; ++c) lse += __expf(logit[c] - m);
  lse = logf(lse);
  #pragma unroll
  for (int c = 0; c < NCLS; ++c) out[(size_t)node * NCLS + c] = logit[c] - m - lse;
}

// ---------------- host side ------------------------------------------------
struct PmaP {
  const float *Wk, *bk, *Wv, *bv, *att, *Wf, *bf, *g0, *b0, *g1, *b1;
};

extern "C" void kernel_launch(void* const* d_in, const int* in_sizes, int n_in,
                              void* d_out, int out_size, void* d_ws, size_t ws_size,
                              hipStream_t stream) {
  const float* node_x  = (const float*)d_in[0];
  const int*   n2e_src = (const int*)d_in[1];
  // d_in[2] = n2e_seg (implicit arange//16), d_in[4] = e2n_seg
  const int*   e2n_src = (const int*)d_in[3];

  auto leaf = [&](int i) { return (const float*)d_in[i]; };
  const float *bn_g, *bn_b, *bn_mean, *bn_var, *Wc, *bc;
  PmaP pma[LL][2];  // [layer][0=n2e, 1=e2n]
  int p = 5;
  const bool sorted = (n_in > 5 && in_sizes[5] == LL * DD * NCLS);  // 2560 -> jax sorted-key order
  if (!sorted) {
    // insertion (dict) order
    bn_g = leaf(p++); bn_b = leaf(p++); bn_mean = leaf(p++); bn_var = leaf(p++);
    for (int l = 0; l < LL; ++l)
      for (int s = 0; s < 2; ++s) {           // n2e then e2n
        PmaP& m = pma[l][s];
        m.Wk = leaf(p++); m.bk = leaf(p++); m.Wv = leaf(p++); m.bv = leaf(p++);
        m.att = leaf(p++); m.Wf = leaf(p++); m.bf = leaf(p++);
        m.g0 = leaf(p++); m.b0 = leaf(p++); m.g1 = leaf(p++); m.b1 = leaf(p++);
      }
    Wc = leaf(p++); bc = leaf(p++);
  } else {
    // jax.tree flatten: dict keys sorted
    Wc = leaf(p++); bc = leaf(p++);
    bn_b = leaf(p++); bn_g = leaf(p++); bn_mean = leaf(p++); bn_var = leaf(p++);
    for (int l = 0; l < LL; ++l)
      for (int s = 1; s >= 0; --s) {          // "e2n" < "n2e"
        PmaP& m = pma[l][s];
        m.Wf = leaf(p++); m.Wk = leaf(p++); m.Wv = leaf(p++); m.att = leaf(p++);
        m.b0 = leaf(p++); m.b1 = leaf(p++); m.bf = leaf(p++);
        m.bk = leaf(p++); m.bv = leaf(p++); m.g0 = leaf(p++); m.g1 = leaf(p++);
      }
  }

  // ---- workspace carve (byte-based) ----
  const size_t rowsz = (size_t)NN * DD;
  char* wsb = (char*)d_ws;
  size_t off = 0;
  auto carve_f32 = [&](void) { float*  q = (float*)(wsb + off);  off += rowsz * 4; return q; };
  auto carve_h   = [&](void) { __bf16* q = (__bf16*)(wsb + off); off += rowsz * 2; return q; };
  float*  Kb   = carve_f32();   // K projections (f32, pooling input)
  float*  Vb   = carve_f32();   // V projections
  float*  Pb   = carve_f32();   // pooled+LN (residual path)
  float*  Fb   = carve_f32();   // FF output
  float*  res0 = carve_f32();   // layer-0 node features (classifier)
  float*  res1 = carve_f32();   // layer-1 node features (classifier)
  __bf16* Xh   = carve_h();     // BN output (GEMM input)
  __bf16* Ph   = carve_h();     // pooled+LN bf16 (FF GEMM input)
  __bf16* EXh  = carve_h();     // edge features bf16 (E2N GEMM input)
  __bf16* r0h  = carve_h();     // layer-0 node features bf16
  __bf16* r1h  = carve_h();     // layer-1 node features bf16
  __bf16* wT[LL][2][3];         // transposed bf16 weights: [layer][pma][Wk,Wv,Wf]
  for (int l = 0; l < LL; ++l)
    for (int s = 0; s < 2; ++s)
      for (int w = 0; w < 3; ++w) { wT[l][s][w] = (__bf16*)(wsb + off); off += (size_t)DD * DD * 2; }

  const dim3 b256(256), b128(128);

  // ---- one-shot weight conversion+transpose (12 matrices) ----
  const int gW = (DD * DD + 255) / 256;
  for (int l = 0; l < LL; ++l)
    for (int s = 0; s < 2; ++s) {
      prep_weight<<<dim3(gW), b256, 0, stream>>>(pma[l][s].Wk, wT[l][s][0]);
      prep_weight<<<dim3(gW), b256, 0, stream>>>(pma[l][s].Wv, wT[l][s][1]);
      prep_weight<<<dim3(gW), b256, 0, stream>>>(pma[l][s].Wf, wT[l][s][2]);
    }

  // ---- input BatchNorm -> bf16 ----
  const int total = NN * DD;
  bn_kernel<<<dim3((total + 255) / 256), b256, 0, stream>>>(
      node_x, bn_mean, bn_var, bn_g, bn_b, Xh, total);

  const __bf16* x = Xh;
  float*  resf[LL] = {res0, res1};
  __bf16* resh[LL] = {r0h, r1h};
  const int gN = (NN + 15) / 16, gE = (EE + 15) / 16;

  for (int l = 0; l < LL; ++l) {
    // ---- N2E: nodes -> hyperedges (E segments of 16 pairs) ----
    const PmaP& pn = pma[l][0];
    gemm128_bf16_wmma<0><<<dim3(gN), b256, 0, stream>>>(x, wT[l][0][0], pn.bk, Kb, NN);
    gemm128_bf16_wmma<0><<<dim3(gN), b256, 0, stream>>>(x, wT[l][0][1], pn.bv, Vb, NN);
    pma_pool_ln<<<dim3(EE), b128, 0, stream>>>(Kb, Vb, n2e_src, pn.att, pn.g0, pn.b0, Pb, Ph, EE);
    gemm128_bf16_wmma<1><<<dim3(gE), b256, 0, stream>>>(Ph, wT[l][0][2], pn.bf, Fb, EE);
    resid_ln_elu<<<dim3(EE), b128, 0, stream>>>(Pb, Fb, pn.g1, pn.b1, nullptr, EXh, EE);

    // ---- E2N: hyperedges -> nodes (N segments of 16 pairs) ----
    const PmaP& pe = pma[l][1];
    gemm128_bf16_wmma<0><<<dim3(gE), b256, 0, stream>>>(EXh, wT[l][1][0], pe.bk, Kb, EE);
    gemm128_bf16_wmma<0><<<dim3(gE), b256, 0, stream>>>(EXh, wT[l][1][1], pe.bv, Vb, EE);
    pma_pool_ln<<<dim3(NN), b128, 0, stream>>>(Kb, Vb, e2n_src, pe.att, pe.g0, pe.b0, Pb, Ph, NN);
    gemm128_bf16_wmma<1><<<dim3(gN), b256, 0, stream>>>(Ph, wT[l][1][2], pe.bf, Fb, NN);
    resid_ln_elu<<<dim3(NN), b128, 0, stream>>>(Pb, Fb, pe.g1, pe.b1, resf[l], resh[l], NN);

    x = resh[l];
  }

  classifier<<<dim3((NN + 127) / 128), b128, 0, stream>>>(
      res0, res1, Wc, bc, (float*)d_out, NN);
}
